// MoEEncoderLayer_3504693313985
// MI455X (gfx1250) — matrix operand order repaired and checked by reading, not compiled, not run on hardware
//
#include <hip/hip_runtime.h>
#include <hip/hip_bf16.h>
#include <stdint.h>

// ---------------------------------------------------------------------------
// MoE encoder layer for MI455X (gfx1250, wave32, WMMA + TDM).
// Heavy math: v_wmma_f32_16x16x32_bf16 (f32 accumulate).
// Staging: Tensor Data Mover (tensor_load_to_lds) double-buffered where
// possible; pipelined global->reg->LDS otherwise.
// ---------------------------------------------------------------------------

typedef __bf16 bf16;
typedef __attribute__((ext_vector_type(8)))  bf16  bf16x8;
typedef __attribute__((ext_vector_type(16))) bf16  v16bf;
typedef __attribute__((ext_vector_type(8)))  float v8f;
typedef __attribute__((ext_vector_type(4)))  float f32x4;
typedef __attribute__((ext_vector_type(4)))  unsigned int u32x4;
typedef __attribute__((ext_vector_type(8)))  int   i32x8;
typedef __attribute__((ext_vector_type(4)))  int   i32x4;
typedef __attribute__((address_space(3)))    void  lds_void;

#define BB 2
#define LL 2048
#define DD 512
#define HH 8
#define HDc 64
#define EE 8
#define DFc 2048
#define TT (BB*LL)

#if __has_builtin(__builtin_amdgcn_tensor_load_to_lds) && \
    __has_builtin(__builtin_amdgcn_s_wait_tensorcnt)
#define HAVE_TDM 1
#else
#define HAVE_TDM 0
#endif

__device__ __forceinline__ float gelu_tanh(float x) {
  const float c = 0.7978845608028654f;
  float x3 = x * x * x;
  return 0.5f * x * (1.0f + tanhf(c * (x + 0.044715f * x3)));
}

#if HAVE_TDM
// Loop-invariant D# group-1 for a (128 x 32) bf16 tile load with LDS row
// padding: pad_interval code 3 = 16 DWORDs (one 64B tile row), pad_amount
// code 3 = 4 DWORDs (16B) -> 80B LDS row pitch matching bf16[128][40].
__device__ __forceinline__ i32x8 tdm_group1(unsigned ld_elems) {
  i32x8 g1 = {0, 0, 0, 0, 0, 0, 0, 0};
  g1[0] = (int)((1u << 16)      // data_size = 2 bytes
              | (1u << 20)      // pad_enable
              | (3u << 22)      // pad_interval = 16 DWORDs
              | (3u << 25));    // pad_amount   = 4 DWORDs
  const unsigned td0 = 1u << 24, td1 = 1u << 24;         // huge dims: no OOB
  g1[1] = (int)((td0 & 0xffffu) << 16);                  // tensor_dim0 lo16
  g1[2] = (int)(((td0 >> 16) & 0xffffu) | ((td1 & 0xffffu) << 16));
  g1[3] = (int)(((td1 >> 16) & 0xffffu) | (32u << 16));  // tile_dim0 = 32
  g1[4] = (int)(128u);                                   // tile_dim1 = 128
  g1[5] = (int)ld_elems;                                 // dim0_stride lo32
  g1[6] = 0;
  return g1;
}

// Issue one 2D TDM load; only group-0 (addresses) varies per call.
__device__ __forceinline__ void tdm_issue(const i32x8& g1, const bf16* gbase,
                                          unsigned lds_byte_off) {
  unsigned long long ga = (unsigned long long)(size_t)gbase;
  u32x4 g0;
  g0[0] = 1u;                                            // count=1 (valid D#)
  g0[1] = lds_byte_off;                                  // lds_addr
  g0[2] = (unsigned)ga;                                  // global_addr lo
  g0[3] = (unsigned)((ga >> 32) & 0x01ffffffu) | (2u << 30); // addr hi | type=2
  i32x4 gz = {0, 0, 0, 0};
#if __has_include(<hip/amd_detail/amd_gfx1250_TDM.h>)
  i32x8 gz8 = {0, 0, 0, 0, 0, 0, 0, 0};
  __builtin_amdgcn_tensor_load_to_lds(g0, g1, gz, gz, gz8, 0);
#else
  __builtin_amdgcn_tensor_load_to_lds(g0, g1, gz, gz, 0);
#endif
}
#endif

enum { EPI_F32 = 0, EPI_QKV = 1, EPI_VT = 2, EPI_CTX = 3, EPI_GELU_BF16 = 4,
       EPI_WACC = 5 };

// ---------------------------------------------------------------------------
// Tiled WMMA GEMM: C[bz] = epi( A[bz] (MxK) * B[bz]^T * scale + bias )
//   B stored transposed: [N][K] row-major.  BM=BN=128, BK=32.
// 256 threads = 8 waves (4x2 wave grid); each wave owns 32x64 = 2x4 WMMA
// accumulators.  Double-buffered LDS; TDM staging when A is bf16.
// ---------------------------------------------------------------------------
template<int EPI, bool A_F32>
__global__ __launch_bounds__(256, 2)
void gemm_wmma(const void* __restrict__ Av, long long strideA, int lda,
               const bf16* __restrict__ Bm, long long strideB, int ldb,
               const float* __restrict__ bias, float scale,
               float* __restrict__ Cf, bf16* __restrict__ Cb,
               long long strideC, int ldc,
               int M, int N, int Kdim,
               const float* __restrict__ wgate, int eidx)
{
  __shared__ __align__(16) bf16 As[2][128][40];
  __shared__ __align__(16) bf16 Bs[2][128][40];

  const int tid  = threadIdx.x;
  const int bm0  = blockIdx.y * 128;
  const int bn0  = blockIdx.x * 128;
  const int bz   = blockIdx.z;
  const int lane = tid & 31;
  const int wave = tid >> 5;
  const int wm   = wave >> 1;     // 0..3  (32-row strip)
  const int wn   = wave & 1;      // 0..1  (64-col strip)
  const int half = lane >> 4;     // wave32 half
  const int lr   = lane & 15;

  const bf16* Bb = Bm + (size_t)bz * (size_t)strideB;

  const v8f vzero = {0.f,0.f,0.f,0.f,0.f,0.f,0.f,0.f};
  v8f acc[2][4];
#pragma unroll
  for (int i = 0; i < 2; ++i)
#pragma unroll
    for (int j = 0; j < 4; ++j) acc[i][j] = vzero;

  const int ntile = Kdim >> 5;

  // per-K-tile compute from LDS buffer `buf`
  auto compute = [&](int buf) {
    v16bf afr[2], bfr[4];
#pragma unroll
    for (int i = 0; i < 2; ++i) {
      int r = wm * 32 + i * 16 + lr;
      // lanes 0-15: K {0..7,16..23}; lanes 16-31: K {8..15,24..31}
      bf16x8 lo = *(const bf16x8*)&As[buf][r][half * 8];
      bf16x8 hi = *(const bf16x8*)&As[buf][r][16 + half * 8];
      afr[i] = __builtin_shufflevector(lo, hi, 0,1,2,3,4,5,6,7,
                                               8,9,10,11,12,13,14,15);
    }
#pragma unroll
    for (int j = 0; j < 4; ++j) {
      int n = wn * 64 + j * 16 + lr;
      bf16x8 lo = *(const bf16x8*)&Bs[buf][n][half * 8];
      bf16x8 hi = *(const bf16x8*)&Bs[buf][n][16 + half * 8];
      bfr[j] = __builtin_shufflevector(lo, hi, 0,1,2,3,4,5,6,7,
                                               8,9,10,11,12,13,14,15);
    }
#pragma unroll
    for (int i = 0; i < 2; ++i)
#pragma unroll
      for (int j = 0; j < 4; ++j)
        acc[i][j] = __builtin_amdgcn_wmma_f32_16x16x32_bf16(
            false, afr[i], false, bfr[j], (short)0, acc[i][j], false, false);
  };

  constexpr bool UseTDM = (HAVE_TDM != 0) && !A_F32;

  if constexpr (UseTDM) {
#if HAVE_TDM
    // ---- TDM-staged, double-buffered main loop (one issuing wave) ----
    const bf16* Abase = (const bf16*)Av + (size_t)bz * strideA +
                        (size_t)bm0 * lda;
    const bf16* Bbase = Bb + (size_t)bn0 * ldb;   // N multiple of 128 here
    const unsigned ldsA[2] = { (unsigned)(size_t)(lds_void*)&As[0][0][0],
                               (unsigned)(size_t)(lds_void*)&As[1][0][0] };
    const unsigned ldsB[2] = { (unsigned)(size_t)(lds_void*)&Bs[0][0][0],
                               (unsigned)(size_t)(lds_void*)&Bs[1][0][0] };
    const i32x8 g1A = tdm_group1((unsigned)lda);
    const i32x8 g1B = tdm_group1((unsigned)ldb);
    if (wave == 0) {
      tdm_issue(g1A, Abase, ldsA[0]);
      tdm_issue(g1B, Bbase, ldsB[0]);
    }
    int buf = 0;
    for (int i = 0; i < ntile; ++i, buf ^= 1) {
      if (wave == 0) __builtin_amdgcn_s_wait_tensorcnt(0);
      __syncthreads();
      if (i + 1 < ntile && wave == 0) {
        int kn = (i + 1) << 5;
        tdm_issue(g1A, Abase + kn, ldsA[buf ^ 1]);
        tdm_issue(g1B, Bbase + kn, ldsB[buf ^ 1]);
      }
      compute(buf);
    }
#endif
  } else {
    // ---- pipelined global->reg->LDS, double-buffered ----
    const int sr = tid >> 2;                 // rows 0..63 (it=0), +64 (it=1)
    const int skc = (tid & 3) << 3;          // 8-elem column chunk

    auto gloadA = [&](int k0, bf16x8 ra[2]) {
#pragma unroll
      for (int it = 0; it < 2; ++it) {
        int r = sr + it * 64;
        if constexpr (!A_F32) {
          const bf16* Ap = (const bf16*)Av + (size_t)bz * strideA +
                           (size_t)(bm0 + r) * lda + k0 + skc;
          ra[it] = *(const bf16x8*)Ap;
        } else {
          const float* Ap = (const float*)Av + (size_t)bz * strideA +
                            (size_t)(bm0 + r) * lda + k0 + skc;
          f32x4 f0 = *(const f32x4*)Ap;
          f32x4 f1 = *(const f32x4*)(Ap + 4);
          bf16x8 vv;
          vv[0] = (bf16)f0[0]; vv[1] = (bf16)f0[1];
          vv[2] = (bf16)f0[2]; vv[3] = (bf16)f0[3];
          vv[4] = (bf16)f1[0]; vv[5] = (bf16)f1[1];
          vv[6] = (bf16)f1[2]; vv[7] = (bf16)f1[3];
          ra[it] = vv;
        }
      }
    };
    auto gloadB = [&](int k0, bf16x8 rb[2]) {
#pragma unroll
      for (int it = 0; it < 2; ++it) {
        int n = sr + it * 64;
        bf16x8 v;
        if (bn0 + n < N) {
          v = *(const bf16x8*)(Bb + (size_t)(bn0 + n) * ldb + k0 + skc);
        } else {
#pragma unroll
          for (int q = 0; q < 8; ++q) v[q] = (bf16)0.f;
        }
        rb[it] = v;
      }
    };
    auto store_lds = [&](int buf, bf16x8 ra[2], bf16x8 rb[2]) {
#pragma unroll
      for (int it = 0; it < 2; ++it) {
        *(bf16x8*)&As[buf][sr + it * 64][skc] = ra[it];
        *(bf16x8*)&Bs[buf][sr + it * 64][skc] = rb[it];
      }
    };

    bf16x8 ra[2], rb[2];
    gloadA(0, ra);
    gloadB(0, rb);
    store_lds(0, ra, rb);
    int buf = 0;
    for (int i = 0; i < ntile; ++i, buf ^= 1) {
      if (i + 1 < ntile) {           // next tile's loads fly during compute
        gloadA((i + 1) << 5, ra);
        gloadB((i + 1) << 5, rb);
      }
      __syncthreads();
      compute(buf);
      if (i + 1 < ntile) store_lds(buf ^ 1, ra, rb);
    }
  }

  // ---- epilogue (C layout: VGPR v -> M = half*8+v, lane lr -> N) ----
#pragma unroll
  for (int i = 0; i < 2; ++i)
#pragma unroll
    for (int j = 0; j < 4; ++j)
#pragma unroll
      for (int v = 0; v < 8; ++v) {
        int m = bm0 + wm * 32 + i * 16 + half * 8 + v;
        int n = bn0 + wn * 64 + j * 16 + lr;
        if (n >= N) continue;
        float val = acc[i][j][v] * scale;
        if (bias) val += bias[n];
        if constexpr (EPI == EPI_F32) {
          Cf[(size_t)bz * strideC + (size_t)m * ldc + n] = val;
        } else if constexpr (EPI == EPI_QKV) {
          int b = m >> 11, l = m & (LL - 1);
          int h = n >> 6,  hd = n & (HDc - 1);
          Cb[((size_t)(b * HH + h) * LL + l) * HDc + hd] = (bf16)val;
        } else if constexpr (EPI == EPI_VT) {
          // V stored transposed [B,H,HD,L] so PV GEMM gets a [N][K] operand
          int b = m >> 11, l = m & (LL - 1);
          int h = n >> 6,  hd = n & (HDc - 1);
          Cb[((size_t)(b * HH + h) * HDc + hd) * LL + l] = (bf16)val;
        } else if constexpr (EPI == EPI_CTX) {
          int b = bz >> 3, h = bz & 7;
          size_t t = (size_t)b * LL + m;
          Cb[t * DD + h * HDc + n] = (bf16)val;
        } else if constexpr (EPI == EPI_GELU_BF16) {
          Cb[(size_t)m * ldc + n] = (bf16)gelu_tanh(val);
        } else { // EPI_WACC: moe[m,n] += w[m,e] * val (experts run sequentially)
          float wv = wgate[m * EE + eidx];
          Cf[(size_t)m * ldc + n] += wv * val;
        }
      }
}

// ---------------------------------------------------------------------------
// Elementwise / reduction helpers
// ---------------------------------------------------------------------------
__global__ void cast_f32_to_bf16(const float* __restrict__ in,
                                 bf16* __restrict__ out, long long n) {
  long long i = (long long)blockIdx.x * blockDim.x + threadIdx.x;
  if (i < n) out[i] = (bf16)in[i];
}

// Cast + transpose: in (R x C f32, slice z) -> out (C x R bf16, slice z).
__global__ __launch_bounds__(256)
void cast_transpose_bf16(const float* __restrict__ in, bf16* __restrict__ out,
                         int R, int C) {
  __shared__ float tile[32][33];
  size_t slab = (size_t)blockIdx.z * (size_t)R * C;
  int c0 = blockIdx.x * 32, r0 = blockIdx.y * 32;
  int tx = threadIdx.x & 31, ty = threadIdx.x >> 5;
#pragma unroll
  for (int i = 0; i < 32; i += 8)
    tile[ty + i][tx] = in[slab + (size_t)(r0 + ty + i) * C + c0 + tx];
  __syncthreads();
#pragma unroll
  for (int i = 0; i < 32; i += 8)
    out[slab + (size_t)(c0 + ty + i) * R + r0 + tx] = (bf16)tile[tx][ty + i];
}

__global__ void fill_zero_f32(float* __restrict__ p, long long n) {
  long long i = (long long)blockIdx.x * blockDim.x + threadIdx.x;
  if (i < n) p[i] = 0.f;
}

__global__ __launch_bounds__(256)
void softmax_row(float* __restrict__ S) {
  __shared__ float red[256];
  float* row = S + (size_t)blockIdx.x * LL;
  int tid = threadIdx.x;
  float mx = -3.4e38f;
  for (int i = tid; i < LL; i += 256) mx = fmaxf(mx, row[i]);
  red[tid] = mx; __syncthreads();
  for (int s = 128; s > 0; s >>= 1) {
    if (tid < s) red[tid] = fmaxf(red[tid], red[tid + s]);
    __syncthreads();
  }
  mx = red[0]; __syncthreads();
  float sum = 0.f;
  for (int i = tid; i < LL; i += 256) {
    float e = __expf(row[i] - mx);
    row[i] = e; sum += e;
  }
  red[tid] = sum; __syncthreads();
  for (int s = 128; s > 0; s >>= 1) {
    if (tid < s) red[tid] += red[tid + s];
    __syncthreads();
  }
  float inv = 1.f / red[0];
  for (int i = tid; i < LL; i += 256) row[i] *= inv;
}

__global__ __launch_bounds__(256)
void ln1_kernel(const float* __restrict__ x, const float* __restrict__ xo,
                const float* __restrict__ g, const float* __restrict__ b,
                float* __restrict__ x1f, bf16* __restrict__ x1b) {
  __shared__ float red[256];
  int t = blockIdx.x, tid = threadIdx.x;
  size_t base = (size_t)t * DD;
  float v0 = x[base + tid]       + xo[base + tid];
  float v1 = x[base + tid + 256] + xo[base + tid + 256];
  red[tid] = v0 + v1; __syncthreads();
  for (int s = 128; s > 0; s >>= 1) { if (tid < s) red[tid] += red[tid + s]; __syncthreads(); }
  float mean = red[0] * (1.f / DD); __syncthreads();
  float d0 = v0 - mean, d1 = v1 - mean;
  red[tid] = d0 * d0 + d1 * d1; __syncthreads();
  for (int s = 128; s > 0; s >>= 1) { if (tid < s) red[tid] += red[tid + s]; __syncthreads(); }
  float rstd = rsqrtf(red[0] * (1.f / DD) + 1e-5f);
  float o0 = d0 * rstd * g[tid]       + b[tid];
  float o1 = d1 * rstd * g[tid + 256] + b[tid + 256];
  x1f[base + tid] = o0;       x1f[base + tid + 256] = o1;
  x1b[base + tid] = (bf16)o0; x1b[base + tid + 256] = (bf16)o1;
}

__global__ __launch_bounds__(256)
void ln_final(const float* __restrict__ gen, const float* __restrict__ moe,
              const float* __restrict__ x1f, const float* __restrict__ g,
              const float* __restrict__ b, float* __restrict__ out,
              float* __restrict__ aug) {
  __shared__ float red[256];
  int t = blockIdx.x, tid = threadIdx.x;
  size_t base = (size_t)t * DD;
  float m0 = (float)(bf16)moe[base + tid];
  float m1 = (float)(bf16)moe[base + tid + 256];
  float v0 = gen[base + tid]       + m0 + x1f[base + tid];
  float v1 = gen[base + tid + 256] + m1 + x1f[base + tid + 256];
  red[tid] = v0 + v1; __syncthreads();
  for (int s = 128; s > 0; s >>= 1) { if (tid < s) red[tid] += red[tid + s]; __syncthreads(); }
  float mean = red[0] * (1.f / DD); __syncthreads();
  float d0 = v0 - mean, d1 = v1 - mean;
  red[tid] = d0 * d0 + d1 * d1; __syncthreads();
  for (int s = 128; s > 0; s >>= 1) { if (tid < s) red[tid] += red[tid + s]; __syncthreads(); }
  float rstd = rsqrtf(red[0] * (1.f / DD) + 1e-5f);
  out[base + tid]       = d0 * rstd * g[tid]       + b[tid];
  out[base + tid + 256] = d1 * rstd * g[tid + 256] + b[tid + 256];
  if (t == 0 && tid == 0) *aug = 0.f;
}

__global__ void gate_logits(const bf16* __restrict__ gh,
                            const float* __restrict__ gW2,
                            const float* __restrict__ gb2,
                            float* __restrict__ logits) {
  int idx = blockIdx.x * blockDim.x + threadIdx.x;
  if (idx >= TT * EE) return;
  int t = idx >> 3, e = idx & 7;
  const bf16* gr = gh + (size_t)t * DD;
  float s = gb2[e];
  for (int d = 0; d < DD; ++d) s += (float)gr[d] * gW2[d * EE + e];
  logits[idx] = s;
}

__global__ void routing_kernel(const float* __restrict__ logits,
                               float* __restrict__ w) {
  int t = blockIdx.x * blockDim.x + threadIdx.x;
  if (t >= TT) return;
  float lg[EE];
#pragma unroll
  for (int e = 0; e < EE; ++e) lg[e] = logits[t * EE + e];
  float mx = lg[0];
#pragma unroll
  for (int e = 1; e < EE; ++e) mx = fmaxf(mx, lg[e]);
  float p[EE], sum = 0.f;
#pragma unroll
  for (int e = 0; e < EE; ++e) { p[e] = __expf(lg[e] - mx); sum += p[e]; }
  float inv = 1.f / sum;
  int i1 = 0;
#pragma unroll
  for (int e = 1; e < EE; ++e) if (lg[e] > lg[i1]) i1 = e;
  int i2 = (i1 == 0) ? 1 : 0;
#pragma unroll
  for (int e = 0; e < EE; ++e) if (e != i1 && lg[e] > lg[i2]) i2 = e;
  float w1 = p[i1] * inv, w2 = p[i2] * inv;
  float ws = w1 + w2 + 1e-9f;
#pragma unroll
  for (int e = 0; e < EE; ++e)
    w[t * EE + e] = (e == i1) ? (w1 / ws) : ((e == i2) ? (w2 / ws) : 0.f);
}

// ---------------------------------------------------------------------------
// Host orchestration
// ---------------------------------------------------------------------------
extern "C" void kernel_launch(void* const* d_in, const int* in_sizes, int n_in,
                              void* d_out, int out_size, void* d_ws, size_t ws_size,
                              hipStream_t stream) {
  (void)in_sizes; (void)n_in; (void)out_size; (void)ws_size;

  const float* x     = (const float*)d_in[0];
  const float* Wq    = (const float*)d_in[1];
  const float* bq    = (const float*)d_in[2];
  const float* Wk    = (const float*)d_in[3];
  const float* bk    = (const float*)d_in[4];
  const float* Wv    = (const float*)d_in[5];
  const float* bv    = (const float*)d_in[6];
  const float* Wo    = (const float*)d_in[7];
  const float* bo    = (const float*)d_in[8];
  const float* ln1_g = (const float*)d_in[9];
  const float* ln1_b = (const float*)d_in[10];
  const float* gW1   = (const float*)d_in[11];
  const float* gb1   = (const float*)d_in[12];
  const float* gW2   = (const float*)d_in[13];
  const float* gb2   = (const float*)d_in[14];
  const float* eW1   = (const float*)d_in[15];
  const float* eb1   = (const float*)d_in[16];
  const float* eW2   = (const float*)d_in[17];
  const float* eb2   = (const float*)d_in[18];
  const float* geW1  = (const float*)d_in[19];
  const float* geb1  = (const float*)d_in[20];
  const float* geW2  = (const float*)d_in[21];
  const float* geb2  = (const float*)d_in[22];
  const float* ln2_g = (const float*)d_in[23];
  const float* ln2_b = (const float*)d_in[24];

  float* outp = (float*)d_out;
  float* attn = outp + (size_t)TT * DD;                  // [B,H,L,L] f32
  float* aug  = attn + (size_t)BB * HH * LL * LL;        // scalar

  size_t off = 0;
  auto alloc = [&](size_t bytes) -> void* {
    void* p = (char*)d_ws + off;
    off += (bytes + 255) & ~(size_t)255;
    return p;
  };
  bf16*  xb    = (bf16*) alloc((size_t)TT * DD * 2);
  bf16*  wqT   = (bf16*) alloc((size_t)DD * DD * 2);      // [N][K]
  bf16*  wkT   = (bf16*) alloc((size_t)DD * DD * 2);
  bf16*  wvT   = (bf16*) alloc((size_t)DD * DD * 2);
  bf16*  woT   = (bf16*) alloc((size_t)DD * DD * 2);
  bf16*  gw1T  = (bf16*) alloc((size_t)DD * DD * 2);
  bf16*  ew1T  = (bf16*) alloc((size_t)EE * DFc * DD * 2); // [E][DF][D]
  bf16*  ew2T  = (bf16*) alloc((size_t)EE * DD * DFc * 2); // [E][D][DF]
  bf16*  gew1T = (bf16*) alloc((size_t)DFc * DD * 2);      // [DF][D]
  bf16*  gew2T = (bf16*) alloc((size_t)DD * DFc * 2);      // [D][DF]
  bf16*  qb    = (bf16*) alloc((size_t)TT * DD * 2);       // [B,H,L,HD]
  bf16*  kb    = (bf16*) alloc((size_t)TT * DD * 2);       // [B,H,L,HD]
  bf16*  vbT   = (bf16*) alloc((size_t)TT * DD * 2);       // [B,H,HD,L]
  bf16*  ctxb  = (bf16*) alloc((size_t)TT * DD * 2);       // [T,D]
  float* xo    = (float*)alloc((size_t)TT * DD * 4);
  float* x1f   = (float*)alloc((size_t)TT * DD * 4);
  bf16*  x1b   = (bf16*) alloc((size_t)TT * DD * 2);
  bf16*  ghb   = (bf16*) alloc((size_t)TT * DD * 2);
  float* logitsf = (float*)alloc((size_t)TT * EE * 4);
  float* wgate   = (float*)alloc((size_t)TT * EE * 4);
  bf16*  hb    = (bf16*) alloc((size_t)TT * DFc * 2);      // reused buffer
  float* moef  = (float*)alloc((size_t)TT * DD * 4);
  float* genf  = (float*)alloc((size_t)TT * DD * 4);

  cast_f32_to_bf16<<<dim3((TT * DD + 255) / 256), 256, 0, stream>>>(
      x, xb, (long long)TT * DD);
  auto castT = [&](const float* s, bf16* dpt, int R, int C, int slices) {
    cast_transpose_bf16<<<dim3(C / 32, R / 32, slices), 256, 0, stream>>>(
        s, dpt, R, C);
  };
  castT(Wq,   wqT,   DD,  DD,  1);
  castT(Wk,   wkT,   DD,  DD,  1);
  castT(Wv,   wvT,   DD,  DD,  1);
  castT(Wo,   woT,   DD,  DD,  1);
  castT(gW1,  gw1T,  DD,  DD,  1);
  castT(eW1,  ew1T,  DD,  DFc, EE);
  castT(eW2,  ew2T,  DFc, DD,  EE);
  castT(geW1, gew1T, DD,  DFc, 1);
  castT(geW2, gew2T, DFc, DD,  1);

  const dim3 blk(256);

  gemm_wmma<EPI_QKV, false><<<dim3(DD/128, TT/128, 1), blk, 0, stream>>>(
      xb, 0, DD, wqT, 0, DD, bq, 1.0f, nullptr, qb, 0, 0, TT, DD, DD, nullptr, 0);
  gemm_wmma<EPI_QKV, false><<<dim3(DD/128, TT/128, 1), blk, 0, stream>>>(
      xb, 0, DD, wkT, 0, DD, bk, 1.0f, nullptr, kb, 0, 0, TT, DD, DD, nullptr, 0);
  gemm_wmma<EPI_VT, false><<<dim3(DD/128, TT/128, 1), blk, 0, stream>>>(
      xb, 0, DD, wvT, 0, DD, bv, 1.0f, nullptr, vbT, 0, 0, TT, DD, DD, nullptr, 0);

  gemm_wmma<EPI_F32, false><<<dim3(LL/128, LL/128, BB*HH), blk, 0, stream>>>(
      qb, (long long)LL * HDc, HDc, kb, (long long)LL * HDc, HDc, nullptr, 0.125f,
      attn, nullptr, (long long)LL * LL, LL, LL, LL, HDc, nullptr, 0);

  softmax_row<<<BB * HH * LL, 256, 0, stream>>>(attn);

  gemm_wmma<EPI_CTX, true><<<dim3(1, LL/128, BB*HH), blk, 0, stream>>>(
      attn, (long long)LL * LL, LL, vbT, (long long)HDc * LL, LL, nullptr, 1.0f,
      nullptr, ctxb, 0, 0, LL, HDc, LL, nullptr, 0);

  gemm_wmma<EPI_F32, false><<<dim3(DD/128, TT/128, 1), blk, 0, stream>>>(
      ctxb, 0, DD, woT, 0, DD, bo, 1.0f, xo, nullptr, 0, DD, TT, DD, DD, nullptr, 0);

  ln1_kernel<<<TT, 256, 0, stream>>>(x, xo, ln1_g, ln1_b, x1f, x1b);

  gemm_wmma<EPI_GELU_BF16, false><<<dim3(DD/128, TT/128, 1), blk, 0, stream>>>(
      x1b, 0, DD, gw1T, 0, DD, gb1, 1.0f, nullptr, ghb, 0, DD, TT, DD, DD, nullptr, 0);
  gate_logits<<<(TT * EE + 255) / 256, 256, 0, stream>>>(ghb, gW2, gb2, logitsf);
  routing_kernel<<<(TT + 255) / 256, 256, 0, stream>>>(logitsf, wgate);

  fill_zero_f32<<<dim3((TT * DD + 255) / 256), 256, 0, stream>>>(
      moef, (long long)TT * DD);
  for (int e = 0; e < EE; ++e) {
    gemm_wmma<EPI_GELU_BF16, false><<<dim3(DFc/128, TT/128, 1), blk, 0, stream>>>(
        x1b, 0, DD, ew1T + (size_t)e * DFc * DD, 0, DD, eb1 + (size_t)e * DFc, 1.0f,
        nullptr, hb, 0, DFc, TT, DFc, DD, nullptr, 0);
    gemm_wmma<EPI_WACC, false><<<dim3(DD/128, TT/128, 1), blk, 0, stream>>>(
        hb, 0, DFc, ew2T + (size_t)e * DD * DFc, 0, DFc, eb2 + (size_t)e * DD, 1.0f,
        moef, nullptr, 0, DD, TT, DD, DFc, wgate, e);
  }

  gemm_wmma<EPI_GELU_BF16, false><<<dim3(DFc/128, TT/128, 1), blk, 0, stream>>>(
      x1b, 0, DD, gew1T, 0, DD, geb1, 1.0f, nullptr, hb, 0, DFc, TT, DFc, DD, nullptr, 0);
  gemm_wmma<EPI_F32, false><<<dim3(DD/128, TT/128, 1), blk, 0, stream>>>(
      hb, 0, DFc, gew2T, 0, DFc, geb2, 1.0f, genf, nullptr, 0, DD, TT, DD, DFc, nullptr, 0);

  ln_final<<<TT, 256, 0, stream>>>(genf, moef, x1f, ln2_g, ln2_b, outp, aug);
}
